// NetVLAD_cent_nonorm_21861383536708
// MI455X (gfx1250) — matrix operand order repaired and checked by reading, not compiled, not run on hardware
//
#include <hip/hip_runtime.h>
#include <hip/hip_bf16.h>
#include <math.h>

typedef __bf16 bf16_t;
typedef bf16_t v16bf __attribute__((ext_vector_type(16)));
typedef float  v8f   __attribute__((ext_vector_type(8)));
typedef unsigned int v8u __attribute__((ext_vector_type(8)));

#define N_      64
#define C_      512
#define HW_     1024
#define K_      64
#define CH      32          // hw chunk width per iteration
#define XHP     40          // [c][h] bf16 row stride: 80B, 16B-aligned, conflict-free
#define XTP     520         // [h][c] bf16 row stride: 1040B, 16B-aligned
#define ALPHA_  50.0f
#define EPS_    1e-12f
#define THREADS 512         // 16 waves

__device__ __forceinline__ unsigned short f2bf_bits(float f) {
  union { float f; unsigned u; } c; c.f = f;
  unsigned r = (c.u + 0x7FFFu + ((c.u >> 16) & 1u)) >> 16;   // round-nearest-even
  return (unsigned short)r;
}
__device__ __forceinline__ float bf2f(unsigned short b) {
  union { unsigned u; float f; } c; c.u = ((unsigned)b) << 16; return c.f;
}
__device__ __forceinline__ v8f vzero8() { v8f z; for (int j = 0; j < 8; ++j) z[j] = 0.f; return z; }

// Fused: normalize -> bf16 LDS tiles -> (feature out || logits WMMA) ->
// softmax -> vlad WMMA (persistent acc, staged B fragments) -> partial flush.
__global__ __launch_bounds__(THREADS)
void nvlad_main(const float* __restrict__ x, const float* __restrict__ Wt /* [C_][K_] */,
                const float* __restrict__ bias, float* __restrict__ sa_out,
                float* __restrict__ feat_out, float* __restrict__ vlad_part,
                float* __restrict__ asum_part, int splits)
{
  __shared__ __align__(16) unsigned short s_wnt[K_ * C_];   // normalized W^T bf16 [k][c]
  __shared__ __align__(16) unsigned short s_xnh[C_ * XHP];  // xn bf16 [c][h]  (vlad B)
  __shared__ __align__(16) unsigned short s_xnt[CH * XTP];  // xn bf16 [h][c]  (logits B, feature)
  __shared__ __align__(16) float          s_logit[K_ * CH];
  __shared__ __align__(16) unsigned short s_a[K_ * CH];     // soft-assign bf16 [k][h]
  __shared__ float s_colss[CH];
  __shared__ float s_colinv[CH];
  __shared__ float s_wss[K_];
  __shared__ float s_asum[K_];
  __shared__ float s_bias[K_];
  __shared__ float s_wrn[K_];

  const int tid   = threadIdx.x;
  const int lane  = tid & 31;
  const int wid   = tid >> 5;            // 16 waves
  const int half  = lane >> 4;
  const int l16   = lane & 15;
  const int n     = blockIdx.x;
  const int split = blockIdx.y;
  const int hwPer = HW_ / splits;
  const int nchunk = hwPer / CH;

  // ---- preamble: bias, W column norms (parallel), normalized W^T bf16 table
  if (tid < K_) { s_bias[tid] = bias[tid]; s_wss[tid] = 0.f; s_asum[tid] = 0.f; }
  __syncthreads();
  {
    int k = tid & (K_ - 1), p = tid >> 6;          // 8 partials per column
    float ss = 0.f;
    for (int c = p * 64; c < p * 64 + 64; ++c) { float v = Wt[c * K_ + k]; ss += v * v; }
    atomicAdd(&s_wss[k], ss);
  }
  __syncthreads();
  if (tid < K_) s_wrn[tid] = 1.f / fmaxf(sqrtf(s_wss[tid]), EPS_);
  __syncthreads();
  for (int e = tid; e < K_ * C_; e += THREADS) {
    int k = e >> 9, c = e & (C_ - 1);
    s_wnt[e] = f2bf_bits(Wt[c * K_ + k] * s_wrn[k]);
  }

  // 16-bit A(16x32) fragment K-offsets (contiguous, matches B): VGPR j -> K = 2j + 16*half
  int kbA[8];
#pragma unroll
  for (int j = 0; j < 8; ++j) kbA[j] = (j << 1) + half * 16;

  const int mtL = wid >> 1, ntL = wid & 1;   // logits tiles (waves 0..7)
  const int mtV = wid & 3,  nbV = wid >> 2;  // vlad: 4 M-tiles x 4 N-quarters (128 cols)

  v8f accv[8];                               // persistent vlad accumulators
#pragma unroll
  for (int t = 0; t < 8; ++t) accv[t] = vzero8();

  const float* xg = x        + (size_t)n * C_ * HW_;
  float* featn    = feat_out + (size_t)n * HW_ * C_;
  float* san      = sa_out   + (size_t)n * K_ * HW_;

  const int h4 = (tid & 7) * 4;              // this thread's 4 hw columns
  const int c0 = tid >> 3;                   // base c row (step 64)

  for (int ch = 0; ch < nchunk; ++ch) {
    const int hw0 = split * hwPer + ch * CH;
    __syncthreads();                          // protect LDS tiles across chunks
    if (tid < CH) s_colss[tid] = 0.f;
    __syncthreads();
    // ---- pass 1: vectorized b128 loads (coalesced) + per-column sum of squares
    float4 xv[8];
    float ss0 = 0.f, ss1 = 0.f, ss2 = 0.f, ss3 = 0.f;
#pragma unroll
    for (int i = 0; i < 8; ++i) {
      int c = c0 + i * 64;
      xv[i] = *(const float4*)(xg + (size_t)c * HW_ + hw0 + h4);
      ss0 += xv[i].x * xv[i].x; ss1 += xv[i].y * xv[i].y;
      ss2 += xv[i].z * xv[i].z; ss3 += xv[i].w * xv[i].w;
    }
    atomicAdd(&s_colss[h4 + 0], ss0); atomicAdd(&s_colss[h4 + 1], ss1);
    atomicAdd(&s_colss[h4 + 2], ss2); atomicAdd(&s_colss[h4 + 3], ss3);
    __syncthreads();
    if (tid < CH) s_colinv[tid] = 1.f / fmaxf(sqrtf(s_colss[tid]), EPS_);
    __syncthreads();
    // ---- pass 2: normalize from registers, bf16 once, both LDS layouts
    const float i0 = s_colinv[h4], i1 = s_colinv[h4 + 1],
                i2 = s_colinv[h4 + 2], i3 = s_colinv[h4 + 3];
#pragma unroll
    for (int i = 0; i < 8; ++i) {
      int c = c0 + i * 64;
      unsigned short b0 = f2bf_bits(xv[i].x * i0), b1 = f2bf_bits(xv[i].y * i1);
      unsigned short b2 = f2bf_bits(xv[i].z * i2), b3 = f2bf_bits(xv[i].w * i3);
      unsigned* dh = (unsigned*)&s_xnh[c * XHP + h4];          // h-pairs contiguous
      dh[0] = (unsigned)b0 | ((unsigned)b1 << 16);
      dh[1] = (unsigned)b2 | ((unsigned)b3 << 16);
      s_xnt[(h4 + 0) * XTP + c] = b0; s_xnt[(h4 + 1) * XTP + c] = b1;
      s_xnt[(h4 + 2) * XTP + c] = b2; s_xnt[(h4 + 3) * XTP + c] = b3;
    }
    __syncthreads();
    if (wid < 8) {
      // ---- logits GEMM (waves 0..7): [64 x 32] = WnT[64 x 512] @ xn[512 x 32]
      v8f acc = vzero8();
      const int rowA = mtL * 16 + l16;
      const int hcol = ntL * 16 + l16;
      const unsigned* wrow = (const unsigned*)&s_wnt[rowA * C_];
      const unsigned* xrow = (const unsigned*)&s_xnt[hcol * XTP];
#pragma unroll
      for (int g = 0; g < 4; ++g) {          // 4 groups x 4 staged k-steps
        v8u au[4], bu[4];
#pragma unroll
        for (int q = 0; q < 4; ++q) {
          int cc = g * 4 + q;
#pragma unroll
          for (int j = 0; j < 8; ++j) au[q][j] = wrow[cc * 16 + (kbA[j] >> 1)];
#pragma unroll
          for (int j = 0; j < 8; ++j) bu[q][j] = xrow[cc * 16 + half * 8 + j];
        }
#pragma unroll
        for (int q = 0; q < 4; ++q)
          acc = __builtin_amdgcn_wmma_f32_16x16x32_bf16(
              false, __builtin_bit_cast(v16bf, au[q]),
              false, __builtin_bit_cast(v16bf, bu[q]),
              (short)0, acc, false, false);
      }
#pragma unroll
      for (int j = 0; j < 8; ++j) {          // D(16x16): M = j + 8*half, N = l16
        int kc = mtL * 16 + j + 8 * half;
        s_logit[kc * CH + hcol] = acc[j];
        san[(size_t)kc * HW_ + hw0 + hcol] = acc[j];   // raw logits (pre-bias)
      }
    } else {
      // ---- feature out (waves 8..15, overlapped with logits WMMA)
      for (int e = tid - 256; e < CH * C_; e += 256) {
        int h = e >> 9, c = e & (C_ - 1);
        featn[(size_t)(hw0 + h) * C_ + c] = bf2f(s_xnt[h * XTP + c]);
      }
    }
    __syncthreads();
    // ---- softmax over K per column (wave 0)
    if (tid < CH) {
      int h = tid;
      float mx = -3.0e38f;
      for (int k = 0; k < K_; ++k)
        mx = fmaxf(mx, ALPHA_ * (s_logit[k * CH + h] + s_bias[k]));
      float sum = 0.f;
      for (int k = 0; k < K_; ++k) {
        float e = __expf(ALPHA_ * (s_logit[k * CH + h] + s_bias[k]) - mx);
        s_logit[k * CH + h] = e;
        sum += e;
      }
      float inv = 1.f / sum;
      for (int k = 0; k < K_; ++k)
        s_a[k * CH + h] = f2bf_bits(s_logit[k * CH + h] * inv);
    }
    __syncthreads();
    // ---- asum accumulation (Σ_h a[k,h])
    if (tid < K_) {
      float s = 0.f;
      for (int h = 0; h < CH; ++h) s += bf2f(s_a[tid * CH + h]);
      s_asum[tid] += s;
    }
    // ---- vlad GEMM: acc[64 x 512] += a[64 x 32] @ xn^T[32 x 512]
    {
      v8u au, bu[8];
      const int rowA = mtV * 16 + l16;
      const unsigned* arow = (const unsigned*)&s_a[rowA * CH];
#pragma unroll
      for (int j = 0; j < 8; ++j) au[j] = arow[kbA[j] >> 1];
      // stage all 8 B fragments (distinct regs -> one dscnt wait, no WAR nops)
#pragma unroll
      for (int t = 0; t < 8; ++t) {
        const unsigned* crow = (const unsigned*)&s_xnh[(nbV * 128 + t * 16 + l16) * XHP];
#pragma unroll
        for (int j = 0; j < 8; ++j) bu[t][j] = crow[half * 8 + j];
      }
      v16bf A = __builtin_bit_cast(v16bf, au);
#pragma unroll
      for (int t = 0; t < 8; ++t)
        accv[t] = __builtin_amdgcn_wmma_f32_16x16x32_bf16(
            false, A, false, __builtin_bit_cast(v16bf, bu[t]),
            (short)0, accv[t], false, false);
    }
  }
  __syncthreads();
  // ---- flush split-K partials with plain stores (f32 atomic BW would dominate)
  float* vp = vlad_part + ((size_t)n * splits + split) * (size_t)K_ * C_;
#pragma unroll
  for (int t = 0; t < 8; ++t) {
    int cN = nbV * 128 + t * 16 + l16;
#pragma unroll
    for (int j = 0; j < 8; ++j) {
      int kc = mtV * 16 + j + 8 * half;
      vp[(size_t)kc * C_ + cN] = accv[t][j];
    }
  }
  if (tid < K_) asum_part[((size_t)n * splits + split) * K_ + tid] = s_asum[tid];
}

// Reduce over splits, subtract (Σa)*W^T (raw W), intra-L2-normalize over C.
__global__ __launch_bounds__(256)
void nvlad_fin(const float* __restrict__ vlad_part, const float* __restrict__ asum_part,
               const float* __restrict__ Wt, float* __restrict__ vlad_out, int splits)
{
  __shared__ float s_red[8];
  const int nk = blockIdx.x;
  const int n = nk >> 6, k = nk & 63;
  const int tid = threadIdx.x;
  float as = 0.f;
  for (int s = 0; s < splits; ++s) as += asum_part[((size_t)n * splits + s) * K_ + k];
  float v[2];
  for (int i = 0; i < 2; ++i) {
    int c = tid + i * 256;
    float acc = 0.f;
    for (int s = 0; s < splits; ++s)
      acc += vlad_part[(((size_t)n * splits + s) * (size_t)K_ + k) * C_ + c];
    v[i] = acc - as * Wt[c * K_ + k];
  }
  float ss = v[0] * v[0] + v[1] * v[1];
  for (int off = 16; off > 0; off >>= 1) ss += __shfl_down(ss, off, 32);
  if ((tid & 31) == 0) s_red[tid >> 5] = ss;
  __syncthreads();
  if (tid == 0) {
    float t = 0.f;
    for (int w = 0; w < 8; ++w) t += s_red[w];
    s_red[0] = 1.f / fmaxf(sqrtf(t), EPS_);
  }
  __syncthreads();
  float sc = s_red[0];
  float* o = vlad_out + ((size_t)n * K_ + k) * C_;
  o[tid] = v[0] * sc;
  o[tid + 256] = v[1] * sc;
}

extern "C" void kernel_launch(void* const* d_in, const int* in_sizes, int n_in,
                              void* d_out, int out_size, void* d_ws, size_t ws_size,
                              hipStream_t stream) {
  const float* x    = (const float*)d_in[0];     // [64,512,32,32]
  const float* Wt   = (const float*)d_in[1];     // [512,64]
  const float* bias = (const float*)d_in[2];     // [64]
  float* out      = (float*)d_out;
  float* vlad_out = out;                                   // [64, 64*512]
  float* sa_out   = out + (size_t)N_ * K_ * C_;            // [64, 64, 1024]
  float* feat_out = sa_out + (size_t)N_ * K_ * HW_;        // [64, 1024, 512]

  // Choose split-K factor that fits the workspace (deterministic per session).
  int splits = 8;
  while (splits > 1 &&
         (size_t)N_ * splits * K_ * (C_ + 1) * sizeof(float) > ws_size)
    splits >>= 1;

  float* vlad_part = (float*)d_ws;                                   // [N,splits,K,C]
  float* asum_part = vlad_part + (size_t)N_ * splits * K_ * C_;      // [N,splits,K]

  dim3 g1(N_, splits);
  nvlad_main<<<g1, dim3(THREADS), 0, stream>>>(x, Wt, bias, sa_out, feat_out,
                                               vlad_part, asum_part, splits);
  nvlad_fin<<<N_ * K_, 256, 0, stream>>>(vlad_part, asum_part, Wt, vlad_out, splits);
}
